// MoE_24867860644521
// MI455X (gfx1250) — matrix operand order repaired and checked by reading, not compiled, not run on hardware
//
#include <hip/hip_runtime.h>
#include <hip/hip_bf16.h>

// ---------------- types ----------------
typedef __attribute__((ext_vector_type(16))) __bf16 v16bf;
typedef __attribute__((ext_vector_type(8)))  float  v8f;
typedef __attribute__((ext_vector_type(4)))  unsigned int u32x4;
typedef __attribute__((ext_vector_type(4)))  float  f32x4;

union BF16Frag { v16bf v; u32x4 q[2]; };

// fp32 -> bf16 (round-to-nearest-even)
__device__ __forceinline__ unsigned short f2bf(float f) {
    unsigned u = __float_as_uint(f);
    unsigned r = 0x7FFFu + ((u >> 16) & 1u);
    return (unsigned short)((u + r) >> 16);
}

// A fragment (16x32 bf16) from row-major [row][k] buffer.
// ISA layout: lane<16 holds row=lane, K {0..7,16..23}; lane>=16 holds row=lane-16, K {8..15,24..31}
__device__ __forceinline__ v16bf loadA(const unsigned short* base, int pitch,
                                       int row, int k, int lane) {
    int lo = lane & 15, hi = lane >> 4;
    const unsigned short* p = base + (size_t)(row + lo) * pitch + k + hi * 8;
    BF16Frag f;
    f.q[0] = *(const u32x4*)(p);
    f.q[1] = *(const u32x4*)(p + 16);
    return f.v;
}

// B fragment (32x16 bf16) from row-major [n][k] weights.
// ISA layout: lane<16 holds col=lane, K {0..7,16..23}; lane>=16 holds col=lane-16, K {8..15,24..31}
__device__ __forceinline__ v16bf loadB(const unsigned short* base, int pitch,
                                       int n, int k, int lane) {
    int lo = lane & 15, hi = lane >> 4;
    const unsigned short* p = base + (size_t)(n + lo) * pitch + k + hi * 8;
    BF16Frag f;
    f.q[0] = *(const u32x4*)(p);
    f.q[1] = *(const u32x4*)(p + 16);
    return f.v;
}

#define WMMA_BF16(a, b, c) \
    __builtin_amdgcn_wmma_f32_16x16x32_bf16(false, (a), false, (b), (short)0, (c), false, false)

// ---------------- prep kernels ----------------

// global average pool: x0[b*C+c] = mean_p x[b][c][p]   (one wave per (b,c))
__global__ __launch_bounds__(256) void moe_pool(const float* __restrict__ x,
                                                float* __restrict__ x0) {
    int idx  = blockIdx.x * 8 + (threadIdx.x >> 5);   // b*C + c, 8192 total
    int lane = threadIdx.x & 31;
    const float* p = x + (size_t)idx * 1024;
    float s = 0.f;
    for (int i = lane; i < 1024; i += 32) s += p[i];
    #pragma unroll
    for (int m = 16; m >= 1; m >>= 1) s += __shfl_xor(s, m, 32);
    if (lane == 0) x0[idx] = s * (1.0f / 1024.0f);
}

// gating: softmax -> top-2 -> renorm softmax. one thread per (g,b)
__global__ __launch_bounds__(256) void moe_gate(const float* __restrict__ x0,
                                                const float* __restrict__ gates,
                                                int* __restrict__ eids,
                                                float* __restrict__ tws) {
    int t = threadIdx.x;            // g*64 + b
    int g = t >> 6, b = t & 63;
    float logit[8];
    #pragma unroll
    for (int e = 0; e < 8; ++e) logit[e] = 0.f;
    for (int c = 0; c < 128; ++c) {
        float xv = x0[b * 128 + c];
        const float* gr = gates + ((size_t)g * 128 + c) * 8;
        #pragma unroll
        for (int e = 0; e < 8; ++e) logit[e] += xv * gr[e];
    }
    float mx = logit[0];
    #pragma unroll
    for (int e = 1; e < 8; ++e) mx = fmaxf(mx, logit[e]);
    float pr[8], se = 0.f;
    #pragma unroll
    for (int e = 0; e < 8; ++e) { pr[e] = expf(logit[e] - mx); se += pr[e]; }
    float inv = 1.f / se;
    #pragma unroll
    for (int e = 0; e < 8; ++e) pr[e] *= inv;
    int i0 = 0;
    #pragma unroll
    for (int e = 1; e < 8; ++e) if (pr[e] > pr[i0]) i0 = e;
    int i1 = (i0 == 0) ? 1 : 0;
    #pragma unroll
    for (int e = 0; e < 8; ++e) if (e != i0 && pr[e] > pr[i1]) i1 = e;
    float d  = expf(pr[i1] - pr[i0]);           // pr[i0] >= pr[i1]
    float s2 = 1.f / (1.f + d);
    eids[t * 2 + 0] = i0; eids[t * 2 + 1] = i1;
    tws [t * 2 + 0] = s2; tws [t * 2 + 1] = d * s2;
}

// fp32 -> bf16 weight conversion
__global__ __launch_bounds__(256) void moe_cvt(const float* __restrict__ src,
                                               unsigned short* __restrict__ dst, int n) {
    int i = blockIdx.x * 256 + threadIdx.x;
    if (i < n) dst[i] = f2bf(src[i]);
}

// ---------------- fused expert kernel ----------------
// grid: (P/64, B, 4 gates), block: 128 threads = 4 waves, 16 pixel-rows per wave.
__global__ __launch_bounds__(128) void moe_main(
    const float* __restrict__ x,            // [B][C=128][P=1024]
    const int* __restrict__ eids,           // [4][B][2]
    const float* __restrict__ tws,          // [4][B][2]
    const unsigned short* __restrict__ W1b, // [E][256][128] bf16
    const unsigned short* __restrict__ W2b, // [E][256][256] bf16
    const unsigned short* __restrict__ W3b, // [E][128][256] bf16
    const float* __restrict__ b1, const float* __restrict__ b2,
    const float* __restrict__ bn_g, const float* __restrict__ bn_b,
    const float* __restrict__ bn_m, const float* __restrict__ bn_v,
    const float* __restrict__ b3,
    float* __restrict__ out)                // [4][B][128][1024]
{
    __shared__ __attribute__((aligned(16))) unsigned short sX[64][136];      // x tile [p][c]
    __shared__ __attribute__((aligned(16))) unsigned short sH[64][264];      // h1 tile [p][h]
    __shared__ __attribute__((aligned(16))) unsigned short sS[4][16][72];    // per-wave h2 shuffle

    const int b    = blockIdx.y;
    const int g    = blockIdx.z;
    const int p0   = blockIdx.x * 64;
    const int tid  = threadIdx.x;
    const int lane = tid & 31;
    const int wave = tid >> 5;
    const int lo   = lane & 15, hi = lane >> 4;
    const int wrow = wave * 16;

    // ---- stage x tile -> LDS bf16 [p][c] (transpose from [c][p]) ----
    {
        const int pq = (tid & 15) * 4;
        for (int c = (tid >> 4); c < 128; c += 8) {
            f32x4 v = *(const f32x4*)(x + ((size_t)b * 128 + c) * 1024 + p0 + pq);
            sX[pq + 0][c] = f2bf(v.x);
            sX[pq + 1][c] = f2bf(v.y);
            sX[pq + 2][c] = f2bf(v.z);
            sX[pq + 3][c] = f2bf(v.w);
        }
    }
    __syncthreads();

    v8f yacc[8];   // persistent 16x128 fp32 output accumulator per wave

    for (int t = 0; t < 2; ++t) {                      // top-2 experts
        const int   e  = eids[((size_t)g * 64 + b) * 2 + t];
        const float tw = tws [((size_t)g * 64 + b) * 2 + t];
        const unsigned short* w1 = W1b + (size_t)e * 256 * 128;
        const unsigned short* w2 = W2b + (size_t)e * 256 * 256;
        const unsigned short* w3 = W3b + (size_t)e * 128 * 256;

        // fold tw*b3[e] into the output accumulator
        #pragma unroll
        for (int j = 0; j < 8; ++j) {
            float bias = tw * b3[e * 128 + j * 16 + lo];
            #pragma unroll
            for (int i = 0; i < 8; ++i) {
                if (t == 0) yacc[j][i] = bias;
                else        yacc[j][i] += bias;
            }
        }

        // ---- GEMM1: h1 = x_tile @ W1[e]^T + b1  (M=16/wave, N=256, K=128) ----
        for (int nc = 0; nc < 256; nc += 64) {
            v8f acc[4];
            #pragma unroll
            for (int j = 0; j < 4; ++j) {
                float bb = b1[e * 256 + nc + j * 16 + lo];
                #pragma unroll
                for (int i = 0; i < 8; ++i) acc[j][i] = bb;
            }
            for (int k = 0; k < 128; k += 32) {
                v16bf a = loadA(&sX[0][0], 136, wrow, k, lane);
                #pragma unroll
                for (int j = 0; j < 4; ++j) {
                    v16bf bm = loadB(w1, 128, nc + j * 16, k, lane);
                    acc[j] = WMMA_BF16(a, bm, acc[j]);
                }
            }
            #pragma unroll
            for (int j = 0; j < 4; ++j)
                #pragma unroll
                for (int i = 0; i < 8; ++i)
                    sH[wrow + i + 8 * hi][nc + j * 16 + lo] = f2bf(acc[j][i]);
        }
        // sH rows are wave-private; per-wave LDS ordering guarantees RAW

        // ---- GEMM2 + BN/ReLU + GEMM3, fused per 64-wide h2 chunk ----
        for (int hc = 0; hc < 256; hc += 64) {
            v8f acc[4];
            #pragma unroll
            for (int j = 0; j < 4; ++j) {
                float bb = b2[e * 256 + hc + j * 16 + lo];
                #pragma unroll
                for (int i = 0; i < 8; ++i) acc[j][i] = bb;
            }
            for (int k = 0; k < 256; k += 32) {
                v16bf a = loadA(&sH[0][0], 264, wrow, k, lane);
                #pragma unroll
                for (int j = 0; j < 4; ++j) {
                    v16bf bm = loadB(w2, 256, hc + j * 16, k, lane);
                    acc[j] = WMMA_BF16(a, bm, acc[j]);
                }
            }
            // BatchNorm (eval) + ReLU + gate-weight prescale -> per-wave shuffle LDS
            #pragma unroll
            for (int j = 0; j < 4; ++j) {
                int   n    = hc + j * 16 + lo;
                float mean = bn_m[e * 256 + n];
                float inv  = bn_g[e * 256 + n] * rsqrtf(bn_v[e * 256 + n] + 1e-5f);
                float beta = bn_b[e * 256 + n];
                #pragma unroll
                for (int i = 0; i < 8; ++i) {
                    float hv = (acc[j][i] - mean) * inv + beta;
                    hv = fmaxf(hv, 0.f) * tw;
                    sS[wave][i + 8 * hi][j * 16 + lo] = f2bf(hv);
                }
            }
            // GEMM3 partial: y += (tw*h2_chunk) @ W3[e]^T   (K-chunk = 64)
            for (int k2 = 0; k2 < 64; k2 += 32) {
                v16bf a = loadA(&sS[wave][0][0], 72, 0, k2, lane);
                #pragma unroll
                for (int j = 0; j < 8; ++j) {
                    v16bf bm = loadB(w3, 256, j * 16, hc + k2, lane);
                    yacc[j] = WMMA_BF16(a, bm, yacc[j]);
                }
            }
        }
    }

    // ---- writeout: [g][b][c][p], 8 consecutive p per lane ----
    float* og = out + ((size_t)g * 64 + b) * 128 * 1024;
    #pragma unroll
    for (int j = 0; j < 8; ++j) {
        int c = j * 16 + lo;
        float* dst = og + (size_t)c * 1024 + p0 + wrow + 8 * hi;
        f32x4 v0 = { yacc[j][0], yacc[j][1], yacc[j][2], yacc[j][3] };
        f32x4 v1 = { yacc[j][4], yacc[j][5], yacc[j][6], yacc[j][7] };
        *(f32x4*)(dst)     = v0;
        *(f32x4*)(dst + 4) = v1;
    }
}

// ---------------- launch ----------------
extern "C" void kernel_launch(void* const* d_in, const int* in_sizes, int n_in,
                              void* d_out, int out_size, void* d_ws, size_t ws_size,
                              hipStream_t stream) {
    const float* x     = (const float*)d_in[0];
    const float* gates = (const float*)d_in[1];
    const float* W1    = (const float*)d_in[2];
    const float* b1    = (const float*)d_in[3];
    const float* W2    = (const float*)d_in[4];
    const float* b2    = (const float*)d_in[5];
    const float* bn_g  = (const float*)d_in[6];
    const float* bn_b  = (const float*)d_in[7];
    const float* bn_m  = (const float*)d_in[8];
    const float* bn_v  = (const float*)d_in[9];
    const float* W3    = (const float*)d_in[10];
    const float* b3    = (const float*)d_in[11];
    float* out = (float*)d_out;

    char* ws = (char*)d_ws;
    float*          x0   = (float*)(ws);                       // 64*128*4   = 32768 B
    int*            eids = (int*)(ws + 32768);                 // 4*64*2*4   =  2048 B
    float*          tws  = (float*)(ws + 32768 + 2048);        //              2048 B
    unsigned short* W1b  = (unsigned short*)(ws + 36864);      // 262144 elems
    unsigned short* W2b  = W1b + 262144;                       // 524288 elems
    unsigned short* W3b  = W2b + 524288;                       // 262144 elems

    moe_pool<<<1024, 256, 0, stream>>>(x, x0);
    moe_gate<<<1, 256, 0, stream>>>(x0, gates, eids, tws);
    moe_cvt<<<(262144 + 255) / 256, 256, 0, stream>>>(W1, W1b, 262144);
    moe_cvt<<<(524288 + 255) / 256, 256, 0, stream>>>(W2, W2b, 524288);
    moe_cvt<<<(262144 + 255) / 256, 256, 0, stream>>>(W3, W3b, 262144);

    dim3 grid(16, 64, 4);   // P/64 tiles, B, 4 gates
    moe_main<<<grid, 128, 0, stream>>>(x, eids, tws, W1b, W2b, W3b,
                                       b1, b2, bn_g, bn_b, bn_m, bn_v, b3, out);
}
